// Model_9191230014035
// MI455X (gfx1250) — matrix operand order repaired and checked by reading, not compiled, not run on hardware
//
#include <hip/hip_runtime.h>
#include <hip/hip_bf16.h>
#include <math.h>

// ---------------------------------------------------------------------------
// HGT forward for MI455X (gfx1250, wave32).
// Dense GEMMs -> v_wmma_f32_16x16x32_bf16, branchless 16x64 tile per wave.
// Edge phase  -> gather/dot + atomic segment softmax + lane-scatter.
// ---------------------------------------------------------------------------

typedef __bf16 bf16_t;
typedef __attribute__((ext_vector_type(8)))  __bf16 v8bf;
typedef __attribute__((ext_vector_type(16))) __bf16 v16bf;
typedef __attribute__((ext_vector_type(8)))  float  v8f;

#define NE    40000
#define NN    15000
#define IND   128
#define HID   256
#define NH    8
#define DD    32
#define NEDGE 300000
#define NLINK 100000
#define SCALE_ATTN 0.17677669529663689f  // 1/sqrt(32)

// ------------------------------- device helpers ---------------------------

__device__ __forceinline__ unsigned fkey(float f) {
  unsigned u = __float_as_uint(f);
  return (u & 0x80000000u) ? ~u : (u | 0x80000000u);
}
__device__ __forceinline__ float funkey(unsigned k) {
  unsigned u = (k & 0x80000000u) ? (k ^ 0x80000000u) : ~k;
  return __uint_as_float(u);
}

// ------------------------------- kernels ----------------------------------

// f32 -> bf16 (mode 0: copy, mode 2: exact GELU then convert)
__global__ __launch_bounds__(256) void k_convert(const float* __restrict__ in,
                                                 bf16_t* __restrict__ out,
                                                 long n, int mode) {
  long i = (long)blockIdx.x * blockDim.x + threadIdx.x;
  if (i >= n) return;
  float v = in[i];
  if (mode == 2) v = 0.5f * v * (1.f + erff(v * 0.70710678118654752f));
  out[i] = (bf16_t)v;
}

// W[K][N] f32 -> Wt[N][K] bf16, batched via blockIdx.z
__global__ __launch_bounds__(256) void k_transpose(const float* __restrict__ W,
                                                   bf16_t* __restrict__ Wt,
                                                   int K, int N, long sIn, long sOut) {
  const float* w = W + (long)blockIdx.z * sIn;
  bf16_t* wt = Wt + (long)blockIdx.z * sOut;
  int i = blockIdx.x * blockDim.x + threadIdx.x;
  if (i >= K * N) return;
  int k = i / N, n = i - k * N;
  wt[(long)n * K + k] = (bf16_t)w[i];
}

// C[M,N] = A[M,K](bf16,row) @ Bt[N,K](bf16,row=B^T) + bias, epilogues fused.
// One wave computes a 16x64 tile; block = 8 waves = 128 rows x 64 cols.
// Out-of-range row/col indices are CLAMPED for loads (stores are guarded),
// so the K inner loop is branchless: pure b128 loads + 4x v_wmma.
__global__ __launch_bounds__(256) void k_gemm(
    const bf16_t* __restrict__ A, int lda, long sA,
    const bf16_t* __restrict__ Bt, int ldbt, long sB,
    const float* __restrict__ bias,
    float* __restrict__ Cf, int ldc,
    bf16_t* __restrict__ Cb, int ldcb, long sC,
    const float* __restrict__ prevX, int ldp,
    const float* __restrict__ skipLogit,
    int M, int N, int K, int act) {
  A  += (long)blockIdx.z * sA;
  Bt += (long)blockIdx.z * sB;
  bf16_t* Cbz = Cb ? (Cb + (long)blockIdx.z * sC) : (bf16_t*)0;

  const int lane  = threadIdx.x & 31;
  const int wave  = threadIdx.x >> 5;
  const int mTile = blockIdx.y * 128 + wave * 16;
  const int nTile = blockIdx.x * 64;
  const int lh = lane >> 4;   // half-wave select
  const int ll = lane & 15;

  const int mrow  = mTile + ll;
  const int kaOff = lh ? 16 : 0;   // A: lanes 16-31 hold K 16..31
  const int kbOff = lh ? 16 : 0;   // B: lanes 16-31 hold K 16..31

  // Clamped base pointers (hoisted v_cndmask; inner loop is branchless).
  const int mclamp = (mrow < M) ? mrow : 0;
  const bf16_t* aRow = A + (long)mclamp * lda + kaOff;
  const bf16_t* bRow0;
  const bf16_t* bRow1;
  const bf16_t* bRow2;
  const bf16_t* bRow3;
  {
    int n0 = nTile + 0  + ll; if (n0 >= N) n0 = 0;
    int n1 = nTile + 16 + ll; if (n1 >= N) n1 = 0;
    int n2 = nTile + 32 + ll; if (n2 >= N) n2 = 0;
    int n3 = nTile + 48 + ll; if (n3 >= N) n3 = 0;
    bRow0 = Bt + (long)n0 * ldbt + kbOff;
    bRow1 = Bt + (long)n1 * ldbt + kbOff;
    bRow2 = Bt + (long)n2 * ldbt + kbOff;
    bRow3 = Bt + (long)n3 * ldbt + kbOff;
  }

  v8f acc0 = {}, acc1 = {}, acc2 = {}, acc3 = {};

#pragma unroll 2
  for (int kb = 0; kb < K; kb += 32) {
    const v8bf* ap = (const v8bf*)(aRow + kb);
    v8bf lo = ap[0];      // K = kaOff + 0..7
    v8bf hi = ap[1];      // K = kaOff + 8..15
    v16bf afrag = __builtin_shufflevector(
        lo, hi, 0,1,2,3,4,5,6,7,8,9,10,11,12,13,14,15);
    v16bf b0 = *(const v16bf*)(bRow0 + kb);
    v16bf b1 = *(const v16bf*)(bRow1 + kb);
    v16bf b2 = *(const v16bf*)(bRow2 + kb);
    v16bf b3 = *(const v16bf*)(bRow3 + kb);
    acc0 = __builtin_amdgcn_wmma_f32_16x16x32_bf16(false, afrag, false, b0,
                                                   (short)0, acc0, false, false);
    acc1 = __builtin_amdgcn_wmma_f32_16x16x32_bf16(false, afrag, false, b1,
                                                   (short)0, acc1, false, false);
    acc2 = __builtin_amdgcn_wmma_f32_16x16x32_bf16(false, afrag, false, b2,
                                                   (short)0, acc2, false, false);
    acc3 = __builtin_amdgcn_wmma_f32_16x16x32_bf16(false, afrag, false, b3,
                                                   (short)0, acc3, false, false);
  }

  float blendA = 1.f, blendB = 0.f;
  if (skipLogit) {
    float s = *skipLogit;
    blendA = 1.f / (1.f + expf(-s));
    blendB = 1.f - blendA;
  }
  const int rBase = mTile + (lh ? 8 : 0);
#define EPILOG(J, ACC)                                                        \
  { int ncol = nTile + (J) * 16 + ll;                                         \
    if (ncol < N) {                                                           \
      float bv = bias ? bias[ncol] : 0.f;                                     \
      _Pragma("unroll")                                                       \
      for (int i = 0; i < 8; i++) {                                           \
        int row = rBase + i;                                                  \
        if (row < M) {                                                        \
          float c = ACC[i] + bv;                                              \
          if (act == 1) c = fmaxf(c, 0.f);                                    \
          if (prevX) c = blendA * c + blendB * prevX[(long)row * ldp + ncol]; \
          if (Cf)  Cf[(long)row * ldc + ncol] = c;                            \
          if (Cbz) Cbz[(long)row * ldcb + ncol] = (bf16_t)c;                  \
        } } } }
  EPILOG(0, acc0) EPILOG(1, acc1) EPILOG(2, acc2) EPILOG(3, acc3)
#undef EPILOG
}

// alpha logits per (edge, head); atomicMax of order-preserving key per (dst, head)
__global__ __launch_bounds__(256) void k_edge_logits(
    const bf16_t* __restrict__ q, const bf16_t* __restrict__ kt,
    const int* __restrict__ src, const int* __restrict__ dst,
    const float* __restrict__ p,
    float* __restrict__ logits, unsigned* __restrict__ mkey, int EH) {
  int idx = blockIdx.x * blockDim.x + threadIdx.x;
  if (idx >= EH) return;
  int e = idx >> 3, h = idx & 7;
  int s = src[e], d = dst[e];
  const v8bf* qv = (const v8bf*)(q  + (long)d * HID + h * DD);
  const v8bf* kv = (const v8bf*)(kt + (long)s * HID + h * DD);
  float acc = 0.f;
#pragma unroll
  for (int t = 0; t < 4; t++) {
    v8bf qa = qv[t], ka = kv[t];
#pragma unroll
    for (int i = 0; i < 8; i++) acc += (float)qa[i] * (float)ka[i];
  }
  float a = acc * p[h] * SCALE_ATTN;
  logits[idx] = a;
  atomicMax(&mkey[d * NH + h], fkey(a));
}

// w = exp(logit - segmax); atomicAdd segment sums
__global__ __launch_bounds__(256) void k_edge_exp(
    float* __restrict__ logits, const int* __restrict__ dst,
    const unsigned* __restrict__ mkey, float* __restrict__ ssum, int EH) {
  int idx = blockIdx.x * blockDim.x + threadIdx.x;
  if (idx >= EH) return;
  int e = idx >> 3, h = idx & 7;
  int d = dst[e];
  float m = funkey(mkey[d * NH + h]);
  if (!(m > -3.0e38f && m < 3.0e38f)) m = 0.f;  // mimic where(isfinite(m), m, 0)
  float w = expf(logits[idx] - m);
  logits[idx] = w;
  atomicAdd(&ssum[d * NH + h], w);
}

// one wave per (edge, head); lane d scatters alpha * vt[src] into agg[dst]
__global__ __launch_bounds__(256) void k_edge_scatter(
    const bf16_t* __restrict__ vt, const int* __restrict__ src,
    const int* __restrict__ dst, const float* __restrict__ w,
    const float* __restrict__ ssum, float* __restrict__ agg, int EH) {
  int gid = blockIdx.x * blockDim.x + threadIdx.x;
  int wid = gid >> 5, lane = gid & 31;
  if (wid >= EH) return;
  int e = wid >> 3, h = wid & 7;
  int s = src[e], d = dst[e];
  float coef = w[wid] / (ssum[d * NH + h] + 1e-16f);
  float v = (float)vt[(long)s * HID + h * DD + lane];
  atomicAdd(&agg[(long)d * HID + h * DD + lane], coef * v);
}

// one wave per link: 256-dim dot via 8 f32/lane + shfl_xor butterfly
__global__ __launch_bounds__(256) void k_score(
    const float* __restrict__ xe, const float* __restrict__ xn,
    const int* __restrict__ ls, const int* __restrict__ ld,
    float* __restrict__ out, int nl) {
  int gid = blockIdx.x * blockDim.x + threadIdx.x;
  int wid = gid >> 5, lane = gid & 31;
  if (wid >= nl) return;
  const float* a = xe + (long)ls[wid] * HID + lane * 8;
  const float* b = xn + (long)ld[wid] * HID + lane * 8;
  float acc = 0.f;
#pragma unroll
  for (int i = 0; i < 8; i++) acc += a[i] * b[i];
#pragma unroll
  for (int off = 16; off > 0; off >>= 1) acc += __shfl_xor(acc, off, 32);
  if (lane == 0) out[wid] = acc;
}

// ------------------------------- host driver ------------------------------
// Input flattening (dict insertion order):
//  0 x_email  1 x_noun
//  2 in.email.w  3 in.email.b  4 in.noun.w  5 in.noun.b
//  per layer l (base = 6 + 24*l):
//   +0..+5  kqv.email {wk,wq,wv,bk,bq,bv}   +6..+11 kqv.noun
//   +12..+14 rel.e2n {a,m,p}  +15..+17 rel.n2e {a,m,p}
//   +18 out.email.w +19 out.email.b +20 out.noun.w +21 out.noun.b
//   +22 skip.email +23 skip.noun
//  54 ei_src 55 ei_dst 56 el_src 57 el_dst

extern "C" void kernel_launch(void* const* d_in, const int* in_sizes, int n_in,
                              void* d_out, int out_size, void* d_ws, size_t ws_size,
                              hipStream_t stream) {
  (void)in_sizes; (void)n_in; (void)out_size; (void)ws_size;
  auto F = [&](int i) { return (const float*)d_in[i]; };

  const float* x_email = F(0);
  const float* x_noun  = F(1);
  const int* ei_src = (const int*)d_in[54];
  const int* ei_dst = (const int*)d_in[55];
  const int* el_src = (const int*)d_in[56];
  const int* el_dst = (const int*)d_in[57];
  float* out = (float*)d_out;

  const int Nn[2] = {NE, NN};

  // -------- workspace carve-out --------
  char* wsb = (char*)d_ws;
  size_t off = 0;
  auto walloc = [&](size_t bytes) -> void* {
    void* p = wsb + off;
    off = (off + bytes + 255) & ~(size_t)255;
    return p;
  };
  float*  xf[2]  = {(float*)walloc((size_t)NE * HID * 4), (float*)walloc((size_t)NN * HID * 4)};
  bf16_t* xb[2]  = {(bf16_t*)walloc((size_t)NE * HID * 2), (bf16_t*)walloc((size_t)NN * HID * 2)};
  bf16_t* inb[2] = {(bf16_t*)walloc((size_t)NE * IND * 2), (bf16_t*)walloc((size_t)NN * IND * 2)};
  bf16_t *qb[2], *kb[2], *vb[2];
  for (int t = 0; t < 2; t++) {
    qb[t] = (bf16_t*)walloc((size_t)Nn[t] * HID * 2);
    kb[t] = (bf16_t*)walloc((size_t)Nn[t] * HID * 2);
    vb[t] = (bf16_t*)walloc((size_t)Nn[t] * HID * 2);
  }
  bf16_t* ktb = (bf16_t*)walloc((size_t)NE * HID * 2);   // reused: kt / gelu(agg_email)
  bf16_t* vtb = (bf16_t*)walloc((size_t)NE * HID * 2);   // reused: vt / gelu(agg_noun)
  float*  agg[2] = {(float*)walloc((size_t)NE * HID * 4), (float*)walloc((size_t)NN * HID * 4)};
  float*  lg   = (float*)walloc((size_t)NEDGE * NH * 4);
  unsigned* mk = (unsigned*)walloc((size_t)NE * NH * 4);
  float*  ssum = (float*)walloc((size_t)NE * NH * 4);
  bf16_t* wscr = (bf16_t*)walloc((size_t)HID * HID * 2); // JIT weight transpose scratch

  // -------- launch helpers --------
  auto conv = [&](const float* in, bf16_t* o, long n, int mode) {
    k_convert<<<dim3((unsigned)((n + 255) / 256)), 256, 0, stream>>>(in, o, n, mode);
  };
  auto tconv = [&](const float* W, bf16_t* Wt, int K, int N, int batch, long sIn, long sOut) {
    k_transpose<<<dim3((K * N + 255) / 256, 1, batch), 256, 0, stream>>>(W, Wt, K, N, sIn, sOut);
  };
  auto gemm = [&](const bf16_t* A, int lda, long sA, const bf16_t* Bt, int ldbt, long sB,
                  const float* bias, float* Cf, int ldc, bf16_t* Cb, int ldcb, long sC,
                  const float* prev, int ldp, const float* skip,
                  int M, int N, int K, int act, int batch) {
    dim3 grid((N + 63) / 64, (M + 127) / 128, batch);
    k_gemm<<<grid, 256, 0, stream>>>(A, lda, sA, Bt, ldbt, sB, bias, Cf, ldc,
                                     Cb, ldcb, sC, prev, ldp, skip, M, N, K, act);
  };

  // -------- input projections: x = relu(x_in @ W + b) --------
  conv(x_email, inb[0], (long)NE * IND, 0);
  conv(x_noun,  inb[1], (long)NN * IND, 0);
  for (int t = 0; t < 2; t++) {
    tconv(F(2 + 2 * t), wscr, IND, HID, 1, 0, 0);
    gemm(inb[t], IND, 0, wscr, IND, 0, F(3 + 2 * t),
         xf[t], HID, xb[t], HID, 0, nullptr, 0, nullptr,
         Nn[t], HID, IND, /*relu*/1, 1);
  }

  // -------- layers --------
  for (int l = 0; l < 2; l++) {
    const int B = 6 + 24 * l;
    // K / Q / V for both node types
    for (int t = 0; t < 2; t++) {
      const int wbase = B + 6 * t;
      tconv(F(wbase + 0), wscr, HID, HID, 1, 0, 0);
      gemm(xb[t], HID, 0, wscr, HID, 0, F(wbase + 3), nullptr, 0,
           kb[t], HID, 0, nullptr, 0, nullptr, Nn[t], HID, HID, 0, 1);
      tconv(F(wbase + 1), wscr, HID, HID, 1, 0, 0);
      gemm(xb[t], HID, 0, wscr, HID, 0, F(wbase + 4), nullptr, 0,
           qb[t], HID, 0, nullptr, 0, nullptr, Nn[t], HID, HID, 0, 1);
      tconv(F(wbase + 2), wscr, HID, HID, 1, 0, 0);
      gemm(xb[t], HID, 0, wscr, HID, 0, F(wbase + 5), nullptr, 0,
           vb[t], HID, 0, nullptr, 0, nullptr, Nn[t], HID, HID, 0, 1);
    }
    // relations: r=0 e2n (src=email, dst=noun), r=1 n2e (src=noun, dst=email)
    for (int r = 0; r < 2; r++) {
      const int st = (r == 0) ? 0 : 1;
      const int dt = 1 - st;
      const int* sidx = (r == 0) ? ei_src : ei_dst;
      const int* didx = (r == 0) ? ei_dst : ei_src;
      const int rbase = B + 12 + 3 * r;
      // kt = k @ a[h], vt = v @ m[h]   (batched over heads)
      tconv(F(rbase + 0), wscr, DD, DD, NH, (long)DD * DD, (long)DD * DD);
      gemm(kb[st], HID, DD, wscr, DD, (long)DD * DD, nullptr, nullptr, 0,
           ktb, HID, DD, nullptr, 0, nullptr, Nn[st], DD, DD, 0, NH);
      tconv(F(rbase + 1), wscr, DD, DD, NH, (long)DD * DD, (long)DD * DD);
      gemm(vb[st], HID, DD, wscr, DD, (long)DD * DD, nullptr, nullptr, 0,
           vtb, HID, DD, nullptr, 0, nullptr, Nn[st], DD, DD, 0, NH);
      // segment softmax + aggregate
      (void)hipMemsetAsync(mk,   0, (size_t)Nn[dt] * NH * 4, stream);
      (void)hipMemsetAsync(ssum, 0, (size_t)Nn[dt] * NH * 4, stream);
      (void)hipMemsetAsync(agg[dt], 0, (size_t)Nn[dt] * HID * 4, stream);
      const int EH = NEDGE * NH;
      k_edge_logits<<<(EH + 255) / 256, 256, 0, stream>>>(
          qb[dt], ktb, sidx, didx, F(rbase + 2), lg, mk, EH);
      k_edge_exp<<<(EH + 255) / 256, 256, 0, stream>>>(lg, didx, mk, ssum, EH);
      k_edge_scatter<<<(EH * 32 + 255) / 256, 256, 0, stream>>>(
          vtb, sidx, didx, lg, ssum, agg[dt], EH);
    }
    // node update: newx = sig(skip) * (gelu(agg) @ Wout + b) + (1-sig) * x
    for (int t = 0; t < 2; t++) {
      bf16_t* gl = (t == 0) ? ktb : vtb;  // kt/vt buffers are dead now
      conv(agg[t], gl, (long)Nn[t] * HID, /*gelu*/2);
      tconv(F(B + 18 + 2 * t), wscr, HID, HID, 1, 0, 0);
      gemm(gl, HID, 0, wscr, HID, 0, F(B + 19 + 2 * t),
           xf[t], HID, xb[t], HID, 0, xf[t], HID, F(B + 22 + t),
           Nn[t], HID, HID, 0, 1);
    }
  }

  // -------- link scorer --------
  k_score<<<(NLINK * 32 + 255) / 256, 256, 0, stream>>>(
      xf[0], xf[1], el_src, el_dst, out, NLINK);
}
